// FourierMMLayer_13657996001425
// MI455X (gfx1250) — matrix operand order repaired and checked by reading, not compiled, not run on hardware
//
#include <hip/hip_runtime.h>

// out[M,768] = x[M,768] @ F[768,768], fp32 in/out, bf16 WMMA compute.
// Memory-bound (~204 MB <-> ~8.8 us @ 23.3 TB/s). Design:
//  - pre-kernel writes F^T as bf16 into d_ws (1.18 MB, L2-resident)
//  - main kernel streams x once, double-buffers LDS tiles,
//    stages B via global_load_async_to_lds_b128 (ASYNCcnt DMA path),
//    converts x->bf16 in-register (v_cvt_pk_bf16_f32), 8 WMMAs/wave/step.

#define DM     768   // d_model (K and N)
#define KB     32    // K step per stage (one bf16 WMMA K)
#define MT     128   // M tile per workgroup
#define NT     128   // N tile per workgroup
#define LDSK   40    // padded LDS row stride in bf16 elements (80B, conflict-free)
#define NSTEPS (DM / KB)

typedef __attribute__((ext_vector_type(16))) __bf16 v16bf;
typedef __attribute__((ext_vector_type(8)))  float  v8f;

union Frag {            // 8 VGPRs = 16 bf16, filled by two 16B LDS loads
    v16bf v;
    uint4 q[2];
};

__device__ __forceinline__ void async_copy_b128(unsigned lds_off, const void* g) {
    // CDNA5 async DMA: memory -> LDS, tracked by ASYNCcnt (ISA 15.18.3 op 98).
    asm volatile("global_load_async_to_lds_b128 %0, %1, off"
                 :: "v"(lds_off), "v"((unsigned long long)(uintptr_t)g)
                 : "memory");
}
__device__ __forceinline__ void wait_async0() {
    asm volatile("s_wait_asynccnt 0" ::: "memory");
}

// ---- one-time: FT[e][d] = bf16(F[d][e]) ----
__global__ __launch_bounds__(256)
void transpose_F_bf16(const float* __restrict__ F, __bf16* __restrict__ FT)
{
    __shared__ float T[32][33];
    const int bx = blockIdx.x * 32;           // e base
    const int by = blockIdx.y * 32;           // d base
    const int tx  = threadIdx.x & 31;
    const int ty4 = (threadIdx.x >> 5) * 4;
    #pragma unroll
    for (int j = 0; j < 4; ++j)
        T[ty4 + j][tx] = F[(size_t)(by + ty4 + j) * DM + bx + tx];
    __syncthreads();
    const int r  = threadIdx.x >> 3;          // e within tile
    const int c4 = (threadIdx.x & 7) * 4;     // d within tile
    union { __bf16 h[4]; uint2 u; } p;
    #pragma unroll
    for (int j = 0; j < 4; ++j) p.h[j] = (__bf16)T[c4 + j][r];
    *(uint2*)(FT + (size_t)(bx + r) * DM + by + c4) = p.u;
}

template <bool USE_FT>
__global__ __launch_bounds__(256)
void fourier_mm_bf16_wmma(const float* __restrict__ x,
                          const float* __restrict__ F,
                          const __bf16* __restrict__ FT,
                          float* __restrict__ out)
{
    __shared__ __bf16 Xs[2][MT * LDSK];   // X tiles, row-major [m][k]
    __shared__ __bf16 Fs[2][NT * LDSK];   // F tiles, transposed [n][k]

    const int tid  = threadIdx.x;
    const int lane = tid & 31;
    const int wid  = tid >> 5;      // 8 waves
    const int lr   = lane & 15;     // row/col within 16
    const int hk   = lane >> 4;     // half-wave selects K-half (ISA 7.12.2)
    const int wm   = wid & 3;       // wave M strip: 4 x 32 rows
    const int wn   = wid >> 2;      // wave N strip: 2 x 64 cols

    const int row0 = blockIdx.y * MT;
    const int col0 = blockIdx.x * NT;

    v8f acc[2][4] = {};

    float4 vx[4];
    auto x_issue = [&](int k0) {           // global loads for an X tile
        #pragma unroll
        for (int it = 0; it < 4; ++it) {
            int idx = tid + it * 256;
            int r = idx >> 3, c4 = (idx & 7) << 2;
            vx[it] = *(const float4*)(x + (size_t)(row0 + r) * DM + k0 + c4);
        }
    };
    auto x_commit = [&](int buf) {         // cvt to bf16, store to LDS
        #pragma unroll
        for (int it = 0; it < 4; ++it) {
            int idx = tid + it * 256;
            int r = idx >> 3, c4 = (idx & 7) << 2;
            union { __bf16 h[4]; uint2 u; } p;
            p.h[0] = (__bf16)vx[it].x; p.h[1] = (__bf16)vx[it].y;
            p.h[2] = (__bf16)vx[it].z; p.h[3] = (__bf16)vx[it].w;
            *(uint2*)&Xs[buf][r * LDSK + c4] = p.u;
        }
    };
    auto f_stage = [&](int buf, int k0) {
        if constexpr (USE_FT) {
            // async DMA 128 rows x 64B of bf16 F^T straight into LDS
            #pragma unroll
            for (int it = 0; it < 2; ++it) {
                int idx = tid + it * 256;       // 0..511 x 16B
                int n = idx >> 2, seg = idx & 3;
                const __bf16* g = FT + (size_t)(col0 + n) * DM + k0 + seg * 8;
                unsigned l = (unsigned)(uintptr_t)&Fs[buf][n * LDSK + seg * 8];
                async_copy_b128(l, g);
            }
        } else {
            // fallback: fp32 load + in-kernel transpose/convert
            #pragma unroll
            for (int it = 0; it < 4; ++it) {
                int idx = tid + it * 256;
                int kk = idx >> 5, n4 = (idx & 31) << 2;
                const float4 v = *(const float4*)(F + (size_t)(k0 + kk) * DM + col0 + n4);
                Fs[buf][(n4 + 0) * LDSK + kk] = (__bf16)v.x;
                Fs[buf][(n4 + 1) * LDSK + kk] = (__bf16)v.y;
                Fs[buf][(n4 + 2) * LDSK + kk] = (__bf16)v.z;
                Fs[buf][(n4 + 3) * LDSK + kk] = (__bf16)v.w;
            }
        }
    };

    // prologue: stage step 0 into buffer 0
    x_issue(0);
    f_stage(0, 0);
    x_commit(0);
    if constexpr (USE_FT) wait_async0();
    __syncthreads();

    for (int s = 0; s < NSTEPS; ++s) {
        const int cur = s & 1, nxt = cur ^ 1;
        const bool more = (s + 1) < NSTEPS;

        // kick off next tile's traffic before touching LDS for this one
        if (more) { x_issue((s + 1) * KB); f_stage(nxt, (s + 1) * KB); }

        // fragment loads per ISA 16-bit A/B lane layouts
        Frag A[2], B[4];
        #pragma unroll
        for (int t = 0; t < 2; ++t) {
            const __bf16* p = &Xs[cur][(wm * 32 + t * 16 + lr) * LDSK + hk * 8];
            A[t].q[0] = *(const uint4*)(p);
            A[t].q[1] = *(const uint4*)(p + 16);
        }
        #pragma unroll
        for (int n = 0; n < 4; ++n) {
            const __bf16* p = &Fs[cur][(wn * 64 + n * 16 + lr) * LDSK + hk * 8];
            B[n].q[0] = *(const uint4*)(p);
            B[n].q[1] = *(const uint4*)(p + 16);
        }

        #pragma unroll
        for (int t = 0; t < 2; ++t)
            #pragma unroll
            for (int n = 0; n < 4; ++n)
                acc[t][n] = __builtin_amdgcn_wmma_f32_16x16x32_bf16(
                    false, A[t].v, false, B[n].v,
                    (short)0, acc[t][n], false, false);

        if (more) {
            x_commit(nxt);
            if constexpr (USE_FT) wait_async0();  // F DMA landed before barrier
        }
        __syncthreads();
    }

    // store: C/D layout: VGPR g -> M = hk*8 + g, N = lr
    #pragma unroll
    for (int t = 0; t < 2; ++t) {
        #pragma unroll
        for (int n = 0; n < 4; ++n) {
            const int col  = col0 + wn * 64 + n * 16 + lr;
            const int rowb = row0 + wm * 32 + t * 16 + hk * 8;
            #pragma unroll
            for (int g = 0; g < 8; ++g)
                out[(size_t)(rowb + g) * DM + col] = acc[t][n][g];
        }
    }
}

extern "C" void kernel_launch(void* const* d_in, const int* in_sizes, int n_in,
                              void* d_out, int out_size, void* d_ws, size_t ws_size,
                              hipStream_t stream) {
    const float* x = (const float*)d_in[0];   // [4, 8192, 768] fp32
    const float* F = (const float*)d_in[1];   // [768, 768] fp32
    float* out     = (float*)d_out;           // [4, 8192, 768] fp32

    const int M = in_sizes[0] / DM;           // 32768 rows
    dim3 grid(DM / NT, M / MT);               // (6, 256) workgroups

    const size_t ft_bytes = (size_t)DM * DM * 2u;
    if (ws_size >= ft_bytes) {
        __bf16* FT = (__bf16*)d_ws;
        transpose_F_bf16<<<dim3(DM / 32, DM / 32), dim3(256), 0, stream>>>(F, FT);
        fourier_mm_bf16_wmma<true><<<grid, dim3(256), 0, stream>>>(x, F, FT, out);
    } else {
        fourier_mm_bf16_wmma<false><<<grid, dim3(256), 0, stream>>>(x, F, nullptr, out);
    }
}
